// gumbel_NS_shGLM_22162031247536
// MI455X (gfx1250) — matrix operand order repaired and checked by reading, not compiled, not run on hardware
//
#include <hip/hip_runtime.h>
#include <hip/hip_bf16.h>
#include <stdint.h>

#define SUBN   20
#define ENO    2000
#define INO    500
#define TDATA  20000
#define TSYN   200
#define THIST  50
#define SBN    3
#define HBN    3
#define SPAD   32      // subunit dim padded to 32 lanes / 2 N-tiles

typedef __attribute__((ext_vector_type(16))) __bf16 v16bf;
typedef __attribute__((ext_vector_type(16))) short  v16s;
typedef __attribute__((ext_vector_type(8)))  float  v8f;

__device__ __forceinline__ unsigned short f2bf(float x) {
    // truncating f32 -> bf16; exact for the 0/1 spike/connection masks
    unsigned u = __builtin_bit_cast(unsigned, x);
    return (unsigned short)(u >> 16);
}

__device__ __forceinline__ float alpha_val(float t, float delta, float tau_exp) {
    float tt = fmaxf(t - delta, 0.0f) / tau_exp;
    return tt * expf(-tt);
}

// ---------------------------------------------------------------------------
// Kernel 1: build bf16-swizzled B matrices, basis kernels, fused C_den*W_sub.
// Sections are disjoint writes; grid-stride loops, no barriers needed.
// ---------------------------------------------------------------------------
__global__ void __launch_bounds__(256)
build_params(const float* __restrict__ C_den,
             const float* __restrict__ C_syn_e, const float* __restrict__ C_syn_i,
             const float* __restrict__ W_syn,   const float* __restrict__ Tau_syn,
             const float* __restrict__ Delta_syn,
             const float* __restrict__ W_hist,  const float* __restrict__ Tau_hist,
             const float* __restrict__ Delta_hist,
             const float* __restrict__ W_sub,   const float* __restrict__ Theta,
             unsigned short* __restrict__ Bsw_e, unsigned short* __restrict__ Bsw_i,
             float* __restrict__ ke_t, float* __restrict__ ki_t,
             float* __restrict__ hlag, float* __restrict__ cdw,
             float* __restrict__ thetap,
             int nch_e, int nch_i)
{
    const int gid  = blockIdx.x * blockDim.x + threadIdx.x;
    const int nthr = gridDim.x * blockDim.x;

    // --- B swizzle: layout [chunk][ntile][lane][16 halves] ---
    // lane l: n = ntile*16 + (l & 15); half j holds K = chunk*32 + (l>>4)*8 + (j<8 ? j : j+8)
    const int tot_e = nch_e * 2 * 32 * 16;
    for (int i = gid; i < tot_e; i += nthr) {
        int j = i & 15, lane = (i >> 4) & 31, nt = (i >> 9) & 1, ch = i >> 10;
        int n = nt * 16 + (lane & 15);
        int k = ch * 32 + ((lane >> 4) << 3) + (j < 8 ? j : j + 8);
        float v = (n < SUBN && k < ENO) ? C_syn_e[n * ENO + k] : 0.0f;
        Bsw_e[i] = f2bf(v);
    }
    const int tot_i = nch_i * 2 * 32 * 16;
    for (int i = gid; i < tot_i; i += nthr) {
        int j = i & 15, lane = (i >> 4) & 31, nt = (i >> 9) & 1, ch = i >> 10;
        int n = nt * 16 + (lane & 15);
        int k = ch * 32 + ((lane >> 4) << 3) + (j < 8 ? j : j + 8);
        float v = (n < SUBN && k < INO) ? C_syn_i[n * INO + k] : 0.0f;
        Bsw_i[i] = f2bf(v);
    }

    // --- synaptic alpha kernels, transposed layout ke_t[k*SPAD + s] ---
    for (int i = gid; i < TSYN * SPAD; i += nthr) {
        int s = i & 31, k = i >> 5;
        float ve = 0.0f, vi = 0.0f;
        if (s < SUBN) {
            float de = expf(Delta_syn[s * 2 + 0]);
            float di = expf(Delta_syn[s * 2 + 1]);
            #pragma unroll
            for (int b = 0; b < SBN; ++b) {
                ve += W_syn[(s * SBN + b) * 2 + 0] *
                      alpha_val((float)k, de, expf(Tau_syn[b * 2 + 0]));
                vi += W_syn[(s * SBN + b) * 2 + 1] *
                      alpha_val((float)k, di, expf(Tau_syn[b * 2 + 1]));
            }
        }
        ke_t[i] = ve;
        ki_t[i] = vi;
    }

    // --- history lag coefficients: hlag[s*64 + d] multiplies y[t-1-d] ---
    for (int i = gid; i < SPAD * 64; i += nthr) {
        int s = i >> 6, d = i & 63;
        float v = 0.0f;
        if (s < SUBN && d < THIST) {
            float dl = Delta_hist[s];
            #pragma unroll
            for (int b = 0; b < HBN; ++b)
                v += W_hist[s * HBN + b] * alpha_val((float)d, dl, expf(Tau_hist[b]));
        }
        hlag[i] = v;
    }

    // --- cdw[s][s2] = C_den[s][s2] * W_sub[s2], padded 32x32 ---
    for (int i = gid; i < SPAD * SPAD; i += nthr) {
        int s = i >> 5, s2 = i & 31;
        cdw[i] = (s < SUBN && s2 < SUBN) ? C_den[s * SUBN + s2] * W_sub[s2] : 0.0f;
    }

    for (int i = gid; i < SPAD; i += nthr)
        thetap[i] = (i < SUBN) ? Theta[i] : 0.0f;
}

// ---------------------------------------------------------------------------
// Kernel 2: WMMA GEMM  out[T][32] = S[T][K] x Csw^T  (bf16 inputs, f32 acc).
// One wave per 16-row M tile; both 16-wide N tiles.
// Main K loop is branch-free (full 32-wide chunks); one wave-uniform
// remainder chunk is handled after the loop with branchless clamped loads.
// ---------------------------------------------------------------------------
__device__ __forceinline__ v16bf pack_a_fast(const float* __restrict__ p) {
    // p points at 24 consecutive in-bounds floats; lane takes 0..7 and 16..23
    float4 f0 = *(const float4*)(p);
    float4 f1 = *(const float4*)(p + 4);
    float4 f2 = *(const float4*)(p + 16);
    float4 f3 = *(const float4*)(p + 20);
    v16s as;
    as[0]  = (short)f2bf(f0.x); as[1]  = (short)f2bf(f0.y);
    as[2]  = (short)f2bf(f0.z); as[3]  = (short)f2bf(f0.w);
    as[4]  = (short)f2bf(f1.x); as[5]  = (short)f2bf(f1.y);
    as[6]  = (short)f2bf(f1.z); as[7]  = (short)f2bf(f1.w);
    as[8]  = (short)f2bf(f2.x); as[9]  = (short)f2bf(f2.y);
    as[10] = (short)f2bf(f2.z); as[11] = (short)f2bf(f2.w);
    as[12] = (short)f2bf(f3.x); as[13] = (short)f2bf(f3.y);
    as[14] = (short)f2bf(f3.z); as[15] = (short)f2bf(f3.w);
    return __builtin_bit_cast(v16bf, as);
}

__global__ void __launch_bounds__(256)
spike_gemm(const float* __restrict__ S, int K, int nch,
           const unsigned short* __restrict__ Bsw,
           float* __restrict__ outp)
{
    const int lane  = threadIdx.x & 31;
    const int wave  = threadIdx.x >> 5;
    const int mtile = blockIdx.x * 8 + wave;
    if (mtile * 16 >= TDATA) return;

    const int   m    = lane & 15;
    const long  row  = (long)mtile * 16 + m;
    const int   cb   = (lane >> 4) ? 8 : 0;   // A half-K base for this lane group
    const float* Srow = S + row * K;

    v8f acc0 = {};
    v8f acc1 = {};

    const int nfull = K >> 5;                 // wave-uniform: full 32-wide chunks

    for (int ch = 0; ch < nfull; ++ch) {
        const int col0 = (ch << 5) + cb;

        // prefetch next chunk of this row into cache
        __builtin_prefetch(Srow + col0 + 32, 0, 1);

        v16bf av = pack_a_fast(Srow + col0);

        v16bf b0 = *(const v16bf*)(Bsw + ((ch * 2 + 0) * 32 + lane) * 16);
        v16bf b1 = *(const v16bf*)(Bsw + ((ch * 2 + 1) * 32 + lane) * 16);

        acc0 = __builtin_amdgcn_wmma_f32_16x16x32_bf16(
                   false, av, false, b0, (short)0, acc0, false, false);
        acc1 = __builtin_amdgcn_wmma_f32_16x16x32_bf16(
                   false, av, false, b1, (short)0, acc1, false, false);
    }

    if (nfull < nch) {                        // wave-uniform remainder chunk
        const int ch   = nfull;
        const int col0 = (ch << 5) + cb;
        v16s as;
        #pragma unroll
        for (int j = 0; j < 16; ++j) {
            int c = col0 + (j < 8 ? j : 8 + j);
            // branchless: clamped (always-valid) load, zero via select
            float v = Srow[min(c, K - 1)];
            as[j] = (short)f2bf(c < K ? v : 0.0f);
        }
        v16bf av = __builtin_bit_cast(v16bf, as);
        v16bf b0 = *(const v16bf*)(Bsw + ((ch * 2 + 0) * 32 + lane) * 16);
        v16bf b1 = *(const v16bf*)(Bsw + ((ch * 2 + 1) * 32 + lane) * 16);
        acc0 = __builtin_amdgcn_wmma_f32_16x16x32_bf16(
                   false, av, false, b0, (short)0, acc0, false, false);
        acc1 = __builtin_amdgcn_wmma_f32_16x16x32_bf16(
                   false, av, false, b1, (short)0, acc1, false, false);
    }

    // C layout: VGPR r -> M = (lane>>4)*8 + r, N = lane&15
    const int mrow0 = mtile * 16 + (lane >> 4) * 8;
    const int n     = lane & 15;
    #pragma unroll
    for (int r = 0; r < 8; ++r) {
        outp[(long)(mrow0 + r) * SPAD + n]      = acc0[r];
        outp[(long)(mrow0 + r) * SPAD + 16 + n] = acc1[r];
    }
}

// ---------------------------------------------------------------------------
// Kernel 3: depthwise causal conv, lane = subunit (coalesced).
// Filter taps staged global->LDS with CDNA5 async-LDS loads (ASYNCcnt),
// no VGPR round trip.
// ---------------------------------------------------------------------------
__global__ void __launch_bounds__(512)
syn_conv(const float* __restrict__ in_e, const float* __restrict__ in_i,
         const float* __restrict__ ke_t, const float* __restrict__ ki_t,
         float* __restrict__ syn)
{
    __shared__ float lke[TSYN * SPAD];
    __shared__ float lki[TSYN * SPAD];

    const int tid = threadIdx.y * 32 + threadIdx.x;   // 0..511
    // async copy: 16 bytes per lane per op, straight into LDS
    for (int i = tid * 4; i < TSYN * SPAD; i += 512 * 4) {
        unsigned            lds_e = (unsigned)(uintptr_t)(&lke[i]);
        unsigned long long  ga_e  = (unsigned long long)(uintptr_t)(ke_t + i);
        asm volatile("global_load_async_to_lds_b128 %0, %1, off"
                     :: "v"(lds_e), "v"(ga_e) : "memory");
        unsigned            lds_i = (unsigned)(uintptr_t)(&lki[i]);
        unsigned long long  ga_i  = (unsigned long long)(uintptr_t)(ki_t + i);
        asm volatile("global_load_async_to_lds_b128 %0, %1, off"
                     :: "v"(lds_i), "v"(ga_i) : "memory");
    }
    asm volatile("s_wait_asynccnt 0" ::: "memory");
    __syncthreads();

    const int s = threadIdx.x;
    const int t = blockIdx.x * 16 + threadIdx.y;
    if (t >= TDATA) return;

    float acc = 0.0f;
    const int kmax = min(TSYN, t + 1);
    for (int k = 0; k < kmax; ++k) {
        acc += lke[k * SPAD + s] * in_e[(t - k) * SPAD + s]
             + lki[k * SPAD + s] * in_i[(t - k) * SPAD + s];
    }
    syn[t * SPAD + s] = acc;
}

// ---------------------------------------------------------------------------
// Kernel 4: sequential tanh recurrence; single wave32, lane = subunit.
// 64-deep ring buffer of past outputs in LDS; taps + C_den row in registers.
// Single-wave workgroup => barriers are free (s_nop); DS is in-order per wave.
// ---------------------------------------------------------------------------
__global__ void __launch_bounds__(32)
scan_glm(const float* __restrict__ syn,  const float* __restrict__ hlag,
         const float* __restrict__ cdw,  const float* __restrict__ thetap,
         const float* __restrict__ W_sub, const float* __restrict__ V_o,
         float* __restrict__ out)
{
    __shared__ float ring[64 * SPAD];
    const int s = threadIdx.x;

    float hl[THIST];
    #pragma unroll
    for (int d = 0; d < THIST; ++d) hl[d] = hlag[s * 64 + d];

    float cd[SUBN];
    #pragma unroll
    for (int j = 0; j < SUBN; ++j) cd[j] = cdw[s * SPAD + j];

    const float th = thetap[s];
    const float w0 = W_sub[0];
    const float vo = V_o[0];

    for (int i = s; i < 64 * SPAD; i += 32) ring[i] = 0.0f;
    __syncthreads();

    for (int t = 0; t < TDATA; ++t) {
        // history term: sum_d hl[d] * y[t-1-d]   (conflict-free: lane==bank)
        float h = 0.0f;
        #pragma unroll
        for (int d = 0; d < THIST; ++d)
            h += hl[d] * ring[((t - 1 - d) & 63) * SPAD + s];

        // propagation: sum_{s2} C_den[s][s2]*W_sub[s2]*y[t-1][s2] (broadcast)
        float prop = 0.0f;
        const int pr = ((t - 1) & 63) * SPAD;
        #pragma unroll
        for (int j = 0; j < SUBN; ++j)
            prop += cd[j] * ring[pr + j];

        float y = tanhf(syn[t * SPAD + s] + prop + th + h);

        __syncthreads();
        ring[(t & 63) * SPAD + s] = y;
        if (s == 0) out[t] = y * w0 + vo;
        __syncthreads();
    }
}

// ---------------------------------------------------------------------------
extern "C" void kernel_launch(void* const* d_in, const int* in_sizes, int n_in,
                              void* d_out, int out_size, void* d_ws, size_t ws_size,
                              hipStream_t stream) {
    const float* S_e        = (const float*)d_in[0];
    const float* S_i        = (const float*)d_in[1];
    const float* C_den      = (const float*)d_in[2];
    const float* C_syn_e    = (const float*)d_in[3];
    const float* C_syn_i    = (const float*)d_in[4];
    const float* W_syn      = (const float*)d_in[5];
    const float* Tau_syn    = (const float*)d_in[6];
    const float* Delta_syn  = (const float*)d_in[7];
    const float* W_hist     = (const float*)d_in[8];
    const float* Tau_hist   = (const float*)d_in[9];
    const float* Delta_hist = (const float*)d_in[10];
    const float* W_sub      = (const float*)d_in[11];
    const float* V_o        = (const float*)d_in[12];
    const float* Theta      = (const float*)d_in[13];
    float* out = (float*)d_out;

    const int nch_e = (ENO + 31) / 32;   // 63
    const int nch_i = (INO + 31) / 32;   // 16

    // --- workspace carve-up (256B aligned) ---
    char* base = (char*)d_ws;
    size_t off = 0;
    auto carve = [&](size_t bytes) {
        char* p = base + off;
        off = (off + bytes + 255) & ~(size_t)255;
        return p;
    };
    unsigned short* Bsw_e = (unsigned short*)carve((size_t)nch_e * 2 * 32 * 16 * 2);
    unsigned short* Bsw_i = (unsigned short*)carve((size_t)nch_i * 2 * 32 * 16 * 2);
    float* ke_t   = (float*)carve((size_t)TSYN * SPAD * 4);
    float* ki_t   = (float*)carve((size_t)TSYN * SPAD * 4);
    float* hlag   = (float*)carve((size_t)SPAD * 64 * 4);
    float* cdw    = (float*)carve((size_t)SPAD * SPAD * 4);
    float* thetap = (float*)carve((size_t)SPAD * 4);
    float* in_e   = (float*)carve((size_t)TDATA * SPAD * 4);
    float* in_i   = (float*)carve((size_t)TDATA * SPAD * 4);
    float* syn    = (float*)carve((size_t)TDATA * SPAD * 4);
    (void)ws_size; (void)in_sizes; (void)n_in; (void)out_size;

    build_params<<<64, 256, 0, stream>>>(
        C_den, C_syn_e, C_syn_i, W_syn, Tau_syn, Delta_syn,
        W_hist, Tau_hist, Delta_hist, W_sub, Theta,
        Bsw_e, Bsw_i, ke_t, ki_t, hlag, cdw, thetap, nch_e, nch_i);

    const int nblk = (TDATA / 16 + 7) / 8;   // 157 blocks of 8 waves
    spike_gemm<<<nblk, 256, 0, stream>>>(S_e, ENO, nch_e, Bsw_e, in_e);
    spike_gemm<<<nblk, 256, 0, stream>>>(S_i, INO, nch_i, Bsw_i, in_i);

    syn_conv<<<TDATA / 16, dim3(32, 16), 0, stream>>>(in_e, in_i, ke_t, ki_t, syn);

    scan_glm<<<1, 32, 0, stream>>>(syn, hlag, cdw, thetap, W_sub, V_o, out);
}
